// MultiHeadAttentionBlock_47081431498924
// MI455X (gfx1250) — compile-verified, matrix operand order checked by reading, not run on hardware
//
#include <hip/hip_runtime.h>

// Problem constants (match reference)
#define Bn   8
#define Sn   1024
#define En   512
#define Hn   8
#define HDn  64
#define CPBH 512

typedef __attribute__((ext_vector_type(16))) _Float16 v16h;
typedef __attribute__((ext_vector_type(8)))  float    v8f;
typedef __attribute__((ext_vector_type(4)))  unsigned int u32x4;
typedef __attribute__((ext_vector_type(8)))  int      i32x8;
typedef __attribute__((ext_vector_type(4)))  int      i32x4;

union Frag16 { v16h v; _Float16 e[16]; uint4 q[2]; };
union Frag32 { v8f  v; float    e[8];  };
union H8     { uint4 u; _Float16 h[8]; };

#if defined(__AMDGCN__) && __has_builtin(__builtin_amdgcn_tensor_load_to_lds) && \
    __has_builtin(__builtin_amdgcn_s_wait_tensorcnt)
#define HAVE_TDM 1
#else
#define HAVE_TDM 0
#endif

// CDNA5 16-bit A/B fragment K-index map (wave32):
//   lanes 0-15 hold K = {0..7, 16..23}, lanes 16-31 hold K = {8..15, 24..31}
__device__ __forceinline__ int kmap16(int lane, int t) {
  int base = (lane & 16) ? 8 : 0;
  return (t < 8) ? (base + t) : (base + 8 + t);
}

// Fragment load: two contiguous 16B runs -> 2x b128 loads (global or LDS).
__device__ __forceinline__ void frag_load(Frag16& f, const _Float16* rowbase, int lane) {
  int kb = (lane & 16) ? 8 : 0;
  f.q[0] = *reinterpret_cast<const uint4*>(rowbase + kb);
  f.q[1] = *reinterpret_cast<const uint4*>(rowbase + kb + 16);
}

__device__ __forceinline__ float red16_max(float x) {
#pragma unroll
  for (int m = 1; m <= 8; m <<= 1) x = fmaxf(x, __shfl_xor(x, m, 32));
  return x;
}
__device__ __forceinline__ float red16_sum(float x) {
#pragma unroll
  for (int m = 1; m <= 8; m <<= 1) x += __shfl_xor(x, m, 32);
  return x;
}
__device__ __forceinline__ float red32_sum(float x) {
#pragma unroll
  for (int m = 1; m <= 16; m <<= 1) x += __shfl_xor(x, m, 32);
  return x;
}

__device__ __forceinline__ v8f wmma_f16(const Frag16& a, const Frag16& b, const Frag32& c) {
  return __builtin_amdgcn_wmma_f32_16x16x32_f16(false, a.v, false, b.v, (short)0, c.v, false, false);
}

#if HAVE_TDM
// TDM 2D tile load: Global -> LDS.  D# per CDNA5 ISA.
// data_size = 2 bytes; dims/strides in elements.  Wave-level op (EXEC ignored).
__device__ __forceinline__ void tdm_load_2d(unsigned lds_off, const void* gaddr,
                                            unsigned tile_w, unsigned tile_h,
                                            unsigned tensor_w, unsigned row_stride) {
  unsigned long long ga = (unsigned long long)(uintptr_t)gaddr;
  u32x4 g0;
  g0[0] = 1u;                                            // count=1, user descriptor
  g0[1] = lds_off;                                       // lds_addr (bytes)
  g0[2] = (unsigned)ga;                                  // global_addr[31:0]
  g0[3] = ((unsigned)(ga >> 32) & 0x1ffffffu) | (2u << 30); // addr[56:32] | type=2
  i32x8 g1;
  g1[0] = 0x00010000;                                    // data_size=1 -> 2 bytes
  g1[1] = (int)((tensor_w & 0xffffu) << 16);             // tensor_dim0[15:0]
  g1[2] = (int)(((tensor_w >> 16) & 0xffffu) |           // tensor_dim0[31:16]
                ((tile_h & 0xffffu) << 16));             // tensor_dim1[15:0]
  g1[3] = (int)((tile_w & 0xffffu) << 16);               // tile_dim0
  g1[4] = (int)(tile_h & 0xffffu);                       // tile_dim1 (tile_dim2=0)
  g1[5] = (int)row_stride;                               // tensor_dim0_stride[31:0]
  g1[6] = 0;
  g1[7] = 0;
  i32x4 z4 = (i32x4)0;
#if __clang_major__ >= 23
  i32x8 z8 = (i32x8)0;
  __builtin_amdgcn_tensor_load_to_lds(g0, g1, z4, z4, z8, 0);
#else
  __builtin_amdgcn_tensor_load_to_lds(g0, g1, z4, z4, 0);
#endif
}
#endif

// ---------------------------------------------------------------------------
// K0: split heads, L2-normalize q/k -> f16 [B,H,S,HD]; V -> f16 TRANSPOSED
// [B,H,HD,S] so PV B-fragments / TDM rows are contiguous.
// ---------------------------------------------------------------------------
__global__ void prep_qkv(const float* __restrict__ q, const float* __restrict__ k,
                         const float* __restrict__ v,
                         _Float16* __restrict__ qn, _Float16* __restrict__ kn,
                         _Float16* __restrict__ vT) {
  int wid  = threadIdx.x >> 5;
  int lane = threadIdx.x & 31;
  long row = (long)blockIdx.x * 8 + wid;          // [0, B*H*S)
  int b = (int)(row / (Hn * Sn));
  int h = (int)((row / Sn) % Hn);
  int s = (int)(row % Sn);
  int bh = b * Hn + h;
  long src = ((long)(b * Sn + s)) * En + h * HDn;
  long dst = ((long)bh * Sn + s) * HDn;

  float e0 = q[src + lane], e1 = q[src + lane + 32];
  float ss = red32_sum(e0 * e0 + e1 * e1);
  float inv = 1.0f / fmaxf(sqrtf(ss), 1e-12f);
  qn[dst + lane]      = (_Float16)(e0 * inv);
  qn[dst + lane + 32] = (_Float16)(e1 * inv);

  e0 = k[src + lane]; e1 = k[src + lane + 32];
  ss = red32_sum(e0 * e0 + e1 * e1);
  inv = 1.0f / fmaxf(sqrtf(ss), 1e-12f);
  kn[dst + lane]      = (_Float16)(e0 * inv);
  kn[dst + lane + 32] = (_Float16)(e1 * inv);

  vT[((long)bh * HDn + lane) * Sn + s]      = (_Float16)v[src + lane];
  vT[((long)bh * HDn + lane + 32) * Sn + s] = (_Float16)v[src + lane + 32];
}

// ---------------------------------------------------------------------------
// K1: w_out -> f16 TRANSPOSED [col][k]; padded cpb_w2 transposed [16][512]
// ---------------------------------------------------------------------------
__global__ void prep_weights(const float* __restrict__ w2, const float* __restrict__ w_out,
                             _Float16* __restrict__ w2pT, _Float16* __restrict__ w16T) {
  int i = blockIdx.x * 256 + threadIdx.x;
  if (i < En * En) {
    int n = i / En, c = i % En;                   // w16T[n][c] = w_out[c][n]
    w16T[i] = (_Float16)w_out[(long)c * En + n];
  }
  if (i < 16 * CPBH) {
    int n = i / CPBH, c = i % CPBH;               // w2pT[n][c] = w2[c][n]
    w2pT[i] = (n < 8) ? (_Float16)w2[c * 8 + n] : (_Float16)0.0f;
  }
}

// ---------------------------------------------------------------------------
// K2: CPB bias.  layer1 (K=2) on VALU, layer2 [S*S,512]@[512,16] via WMMA.
// Bias stored f16 TRANSPOSED [H][key j][query i].
// ---------------------------------------------------------------------------
__global__ void cpb_bias(const float* __restrict__ rel_x, const float* __restrict__ rel_y,
                         const float* __restrict__ w1, const float* __restrict__ b1,
                         const _Float16* __restrict__ w2pT, _Float16* __restrict__ biasT) {
  __shared__ alignas(16) float w1s[2 * CPBH];
  __shared__ alignas(16) float b1s[CPBH];
  for (int i = threadIdx.x; i < 2 * CPBH; i += 256) w1s[i] = w1[i];
  for (int i = threadIdx.x; i < CPBH;     i += 256) b1s[i] = b1[i];
  __syncthreads();

  int wid  = threadIdx.x >> 5;
  int lane = threadIdx.x & 31;
  int ln   = lane & 15;
  long tile = (long)blockIdx.x * 8 + wid;          // 16 pairs per tile
  long p    = tile * 16 + ln;                      // flat pair idx (i*S + j)
  float rx = rel_x[p], ry = rel_y[p];

  Frag32 acc;
#pragma unroll
  for (int i = 0; i < 8; ++i) acc.e[i] = 0.0f;

  for (int kc = 0; kc < CPBH / 32; ++kc) {
    Frag16 a, bfr;
#pragma unroll
    for (int t = 0; t < 16; ++t) {
      int c = kc * 32 + kmap16(lane, t);
      float hv = fmaf(rx, w1s[c], fmaf(ry, w1s[CPBH + c], b1s[c]));
      a.e[t] = (_Float16)fmaxf(hv, 0.0f);
    }
    frag_load(bfr, w2pT + ln * CPBH + kc * 32, lane);   // direct from global (L2)
    acc.v = wmma_f16(a, bfr, acc);
  }

  if (ln < 8) {                                    // only 8 heads valid
#pragma unroll
    for (int r = 0; r < 8; ++r) {
      int rowM = (lane < 16) ? r : (r + 8);
      long pr  = tile * 16 + rowM;                 // pr = i*S + j
      long i_q = pr >> 10, j_k = pr & (Sn - 1);
      biasT[(long)ln * Sn * Sn + j_k * Sn + i_q] = (_Float16)acc.e[r];
    }
  }
}

// ---------------------------------------------------------------------------
// K3: flash attention.  K/V tiles double-buffered into LDS by the Tensor Data
// Mover (wave 0 issues; s_wait_tensorcnt + barrier hand off), overlapping DMA
// of tile j+1 with WMMA on tile j.  Q/bias fragments direct from global.
// ---------------------------------------------------------------------------
__global__ void flash_attn(const _Float16* __restrict__ qn, const _Float16* __restrict__ kn,
                           const _Float16* __restrict__ vT, const _Float16* __restrict__ biasT,
                           const float* __restrict__ logit_scale,
                           _Float16* __restrict__ vals16) {
  __shared__ alignas(16) _Float16 k_buf[2][32 * 64];   // [col][hd]
  __shared__ alignas(16) _Float16 v_buf[2][64 * 32];   // [hd][col]
  __shared__ alignas(16) _Float16 p_t[8 * 16 * 32];

  int bh      = blockIdx.x >> 3;                   // b*H + h
  int mt      = blockIdx.x & 7;
  int h       = bh & (Hn - 1);
  int b       = bh >> 3;
  int rowbase = mt * 128;
  long hoff   = (long)bh * Sn * HDn;               // also the [B,H,HD,S] head offset

  int wid  = threadIdx.x >> 5;
  int lane = threadIdx.x & 31;
  int ln   = lane & 15;

  float scale = __expf(fminf(logit_scale[h], 4.6051701860f)); // ln(100)

  int rowQ = rowbase + wid * 16 + ln;
  Frag16 aq0, aq1;                                 // Q fragments, K = 0..31 / 32..63
  frag_load(aq0, qn + hoff + (long)rowQ * HDn, lane);
  frag_load(aq1, qn + hoff + (long)rowQ * HDn + 32, lane);

  const _Float16* biasH = biasT + (long)h * Sn * Sn;
  int rsel = rowbase + wid * 16 + ((lane & 16) ? 8 : 0);   // this lane's 8-row run

  float mrow[8], lrow[8];
  Frag32 o[4];
#pragma unroll
  for (int r = 0; r < 8; ++r) { mrow[r] = -1e30f; lrow[r] = 0.0f; }
#pragma unroll
  for (int n0 = 0; n0 < 4; ++n0)
#pragma unroll
    for (int r = 0; r < 8; ++r) o[n0].e[r] = 0.0f;

  _Float16* pw = &p_t[wid * 16 * 32];              // per-wave private region

#if HAVE_TDM
  if (wid == 0) {                                  // prologue: DMA tile 0
    tdm_load_2d((unsigned)(uintptr_t)&k_buf[0][0], kn + hoff, 2048, 1, 2048, 2048);
    tdm_load_2d((unsigned)(uintptr_t)&v_buf[0][0], vT + hoff, 32, 64, Sn, Sn);
  }
#endif

  for (int jt = 0; jt < Sn / 32; ++jt) {
    int col0 = jt * 32;
    int cur  = jt & 1;

#if HAVE_TDM
    if (wid == 0 && jt + 1 < Sn / 32) {            // DMA next tile into other buffer
      int nc0 = (jt + 1) * 32;
      tdm_load_2d((unsigned)(uintptr_t)&k_buf[cur ^ 1][0],
                  kn + hoff + (long)nc0 * HDn, 2048, 1, 2048, 2048);
      tdm_load_2d((unsigned)(uintptr_t)&v_buf[cur ^ 1][0],
                  vT + hoff + nc0, 32, 64, Sn, Sn);
    }
    // in-order TDM: cnt<=2 means the older pair (current tile) completed
    if (jt + 1 < Sn / 32) __builtin_amdgcn_s_wait_tensorcnt(2);
    else                  __builtin_amdgcn_s_wait_tensorcnt(0);
    __syncthreads();
#else
    {                                              // cooperative fallback staging
      uint4* kd = reinterpret_cast<uint4*>(&k_buf[cur][0]);
      const uint4* ks = reinterpret_cast<const uint4*>(kn + hoff + (long)col0 * HDn);
      kd[threadIdx.x] = ks[threadIdx.x];           // 256 x 16B = 4KB
      int r = threadIdx.x >> 2, c8 = threadIdx.x & 3;
      *reinterpret_cast<uint4*>(&v_buf[cur][r * 32 + c8 * 8]) =
          *reinterpret_cast<const uint4*>(vT + hoff + (long)r * Sn + col0 + c8 * 8);
    }
    __syncthreads();
#endif

    float s0[8], s1[8];
#pragma unroll
    for (int sub = 0; sub < 2; ++sub) {
      Frag16 bk0, bk1;
      frag_load(bk0, &k_buf[cur][(sub * 16 + ln) * 64], lane);
      frag_load(bk1, &k_buf[cur][(sub * 16 + ln) * 64 + 32], lane);
      Frag32 acc;
#pragma unroll
      for (int r = 0; r < 8; ++r) acc.e[r] = 0.0f;
      acc.v = wmma_f16(aq0, bk0, acc);
      acc.v = wmma_f16(aq1, bk1, acc);
      H8 bb;                                       // 8 bias values along query rows
      bb.u = *reinterpret_cast<const uint4*>(
          biasH + (long)(col0 + sub * 16 + ln) * Sn + rsel);
#pragma unroll
      for (int r = 0; r < 8; ++r) {
        float sv = acc.e[r] * scale + (float)bb.h[r];
        if (sub == 0) s0[r] = sv; else s1[r] = sv;
      }
    }

    // online softmax update (rows live across 16-lane groups)
#pragma unroll
    for (int r = 0; r < 8; ++r) {
      float tm    = red16_max(fmaxf(s0[r], s1[r]));
      float mnew  = fmaxf(mrow[r], tm);
      float alpha = __expf(mrow[r] - mnew);
      float p0    = __expf(s0[r] - mnew);
      float p1    = __expf(s1[r] - mnew);
      lrow[r] = lrow[r] * alpha + red16_sum(p0 + p1);
      mrow[r] = mnew;
#pragma unroll
      for (int n0 = 0; n0 < 4; ++n0) o[n0].e[r] *= alpha;
      s0[r] = p0; s1[r] = p1;
    }

    // transpose P (C-layout -> A-layout) through per-wave LDS region
#pragma unroll
    for (int r = 0; r < 8; ++r) {
      int rowM = (lane < 16) ? r : (r + 8);
      pw[rowM * 32 + ln]      = (_Float16)s0[r];
      pw[rowM * 32 + 16 + ln] = (_Float16)s1[r];
    }
    Frag16 ap;
    frag_load(ap, pw + ln * 32, lane);

#pragma unroll
    for (int n0 = 0; n0 < 4; ++n0) {
      Frag16 bv;
      frag_load(bv, &v_buf[cur][(n0 * 16 + ln) * 32], lane);
      o[n0].v = wmma_f16(ap, bv, o[n0]);
    }
    __syncthreads();                               // release buf before next DMA
  }

#pragma unroll
  for (int r = 0; r < 8; ++r) {
    int rowM = (lane < 16) ? r : (r + 8);
    long row = rowbase + wid * 16 + rowM;
    float invl = 1.0f / fmaxf(lrow[r], 1e-30f);
#pragma unroll
    for (int n0 = 0; n0 < 4; ++n0)
      vals16[((long)b * Sn + row) * En + h * HDn + n0 * 16 + ln] =
          (_Float16)(o[n0].e[r] * invl);
  }
}

// ---------------------------------------------------------------------------
// K4: output projection: out[8192,512] = vals16 @ w_out + b_out  (f32 out)
// w_out pre-transposed -> both fragment streams load straight from global.
// ---------------------------------------------------------------------------
__global__ void out_proj(const _Float16* __restrict__ vals16, const _Float16* __restrict__ w16T,
                         const float* __restrict__ b_out, float* __restrict__ out) {
  int rb = blockIdx.x >> 3, cb = blockIdx.x & 7;
  int mbase = rb * 128, nbase = cb * 64;
  int wid = threadIdx.x >> 5, lane = threadIdx.x & 31;
  int ln = lane & 15;

  const _Float16* arow = vals16 + (long)(mbase + wid * 16 + ln) * En;

  Frag32 acc[4];
#pragma unroll
  for (int n0 = 0; n0 < 4; ++n0)
#pragma unroll
    for (int r = 0; r < 8; ++r) acc[n0].e[r] = 0.0f;

  for (int kc = 0; kc < En / 32; ++kc) {
    Frag16 a;
    frag_load(a, arow + kc * 32, lane);
#pragma unroll
    for (int n0 = 0; n0 < 4; ++n0) {
      Frag16 bb;
      frag_load(bb, w16T + (long)(nbase + n0 * 16 + ln) * En + kc * 32, lane);
      acc[n0].v = wmma_f16(a, bb, acc[n0]);
    }
  }

#pragma unroll
  for (int r = 0; r < 8; ++r) {
    int rowM = (lane < 16) ? r : (r + 8);
    long row = mbase + wid * 16 + rowM;
#pragma unroll
    for (int n0 = 0; n0 < 4; ++n0) {
      int col = nbase + n0 * 16 + ln;
      out[row * En + col] = acc[n0].e[r] + b_out[col];
    }
  }
}

// ---------------------------------------------------------------------------
extern "C" void kernel_launch(void* const* d_in, const int* in_sizes, int n_in,
                              void* d_out, int out_size, void* d_ws, size_t ws_size,
                              hipStream_t stream) {
  const float* q     = (const float*)d_in[0];
  const float* k     = (const float*)d_in[1];
  const float* v     = (const float*)d_in[2];
  const float* rel_x = (const float*)d_in[3];
  const float* rel_y = (const float*)d_in[4];
  const float* w1    = (const float*)d_in[5];
  const float* b1    = (const float*)d_in[6];
  const float* w2    = (const float*)d_in[7];
  const float* ls    = (const float*)d_in[8];
  const float* w_out = (const float*)d_in[9];
  const float* bo    = (const float*)d_in[10];

  char* ws = (char*)d_ws;
  size_t off = 0;
  _Float16* qn     = (_Float16*)(ws + off); off += (size_t)Bn * Hn * Sn * HDn * 2; // 8 MB
  _Float16* kn     = (_Float16*)(ws + off); off += (size_t)Bn * Hn * Sn * HDn * 2; // 8 MB
  _Float16* vT     = (_Float16*)(ws + off); off += (size_t)Bn * Hn * Sn * HDn * 2; // 8 MB  [B,H,HD,S]
  _Float16* biasT  = (_Float16*)(ws + off); off += (size_t)Hn * Sn * Sn * 2;       // 16 MB [H,key,query]
  _Float16* vals16 = (_Float16*)(ws + off); off += (size_t)Bn * Sn * En * 2;       // 8 MB
  _Float16* w16T   = (_Float16*)(ws + off); off += (size_t)En * En * 2;            // 0.5 MB [col][k]
  _Float16* w2pT   = (_Float16*)(ws + off); off += (size_t)16 * CPBH * 2;          // 16 KB [16][512]

  prep_qkv<<<Bn * Hn * Sn / 8, 256, 0, stream>>>(q, k, v, qn, kn, vT);
  prep_weights<<<(En * En + 255) / 256, 256, 0, stream>>>(w2, w_out, w2pT, w16T);
  cpb_bias<<<(Sn * Sn / 16) / 8, 256, 0, stream>>>(rel_x, rel_y, w1, b1, w2pT, biasT);
  flash_attn<<<Bn * Hn * (Sn / 128), 256, 0, stream>>>(qn, kn, vT, biasT, ls, vals16);
  out_proj<<<(Bn * Sn / 128) * (En / 64), 256, 0, stream>>>(vals16, w16T, bo, (float*)d_out);
}
